// HAN_52467320487955
// MI455X (gfx1250) — matrix operand order repaired and checked by reading, not compiled, not run on hardware
//
#include <hip/hip_runtime.h>

#define DEVINL __device__ __forceinline__

typedef __attribute__((ext_vector_type(16))) __bf16          bf16x16;
typedef __attribute__((ext_vector_type(16))) unsigned short  ushort16;
typedef __attribute__((ext_vector_type(8)))  float           floatx8;

// ---------- helpers ----------

DEVINL unsigned short f2bf(float f) {                 // RNE float -> bf16 bits
  unsigned u = __float_as_uint(f);
  u += 0x7FFFu + ((u >> 16) & 1u);
  return (unsigned short)(u >> 16);
}

DEVINL void atomic_add_f32(float* p, float v) {       // HW f32 atomic, no CAS loop
  asm volatile("global_atomic_add_f32 %0, %1, off" :: "v"(p), "v"(v) : "memory");
}

DEVINL void atomic_max_f32(float* p, float v) {       // classic int-trick float max
  if (v >= 0.0f) atomicMax((int*)p,          __float_as_int(v));
  else           atomicMin((unsigned int*)p, __float_as_uint(v));
}

// ---------- weight pre-pack into WMMA B-matrix lane layout ----------
// B (32x16 bf16 per K-tile/N-tile): lane<16 -> col=lane,    K = kt*32 + {0..7,16..23}
//                                   lane>=16 -> col=lane-16, K = kt*32 + {8..15,24..31}
// stored as [ (kt*4+nt)*32 + lane ][16] contiguous bf16 per lane.
__global__ void pack_w_kernel(const float* __restrict__ W, unsigned short* __restrict__ out) {
  int b = blockIdx.x, lane = threadIdx.x;
  int kt = b >> 2, nt = b & 3;
  int lhalf = lane >> 4, col = lane & 15;
  size_t base = ((size_t)b * 32 + lane) * 16;
#pragma unroll
  for (int i = 0; i < 16; ++i) {
    int k = kt * 32 + lhalf * 8 + (i < 8 ? i : 8 + i);
    out[base + i] = f2bf(W[(size_t)k * 64 + nt * 16 + col]);
  }
}

// ---------- GEMM: Y[N,64] = act(X[N,K] @ W + b), one wave per 16-row strip ----------
template<int K, int ACT>   // ACT: 0=none, 1=ELU
__global__ void gemm_bias_act(const float* __restrict__ X,
                              const unsigned short* __restrict__ Wp,
                              const float* __restrict__ bias,
                              float* __restrict__ Y, int N) {
  int wave = threadIdx.x >> 5, lane = threadIdx.x & 31;
  int rowbase = (blockIdx.x * (blockDim.x >> 5) + wave) * 16;
  if (rowbase >= N) return;
  int lhalf = lane >> 4, lrow = lane & 15;
  int r = rowbase + lrow;
  int rc = r < N ? r : N - 1;
  // A fragment source: lane<16 needs K {0..7,16..23}, lane>=16 needs {8..15,24..31} per K-chunk
  const float* xrow = X + (size_t)rc * K + lhalf * 8;
  const bf16x16* Wv = (const bf16x16*)Wp;
  floatx8 acc[4] = {};
#pragma unroll
  for (int kt = 0; kt < K / 32; ++kt) {
    const float* xp = xrow + kt * 32;
    float4 c0 = *(const float4*)(xp);
    float4 c1 = *(const float4*)(xp + 4);
    float4 c2 = *(const float4*)(xp + 16);
    float4 c3 = *(const float4*)(xp + 20);
    float f[16] = {c0.x,c0.y,c0.z,c0.w, c1.x,c1.y,c1.z,c1.w,
                   c2.x,c2.y,c2.z,c2.w, c3.x,c3.y,c3.z,c3.w};
    ushort16 u;
#pragma unroll
    for (int i = 0; i < 16; ++i) u[i] = f2bf(f[i]);
    bf16x16 a = __builtin_bit_cast(bf16x16, u);
#pragma unroll
    for (int nt = 0; nt < 4; ++nt) {
      bf16x16 bm = Wv[(size_t)(kt * 4 + nt) * 32 + lane];
      acc[nt] = __builtin_amdgcn_wmma_f32_16x16x32_bf16(
          false, a, false, bm, (short)0, acc[nt], false, false);
    }
  }
  // C/D layout: col = lane%16, row = v + 8*(lane/16)
#pragma unroll
  for (int nt = 0; nt < 4; ++nt) {
    int col = nt * 16 + lrow;
    float bv = bias[col];
#pragma unroll
    for (int v = 0; v < 8; ++v) {
      int row = rowbase + lhalf * 8 + v;
      if (row < N) {
        float val = acc[nt][v] + bv;
        if (ACT == 1) val = val > 0.0f ? val : (expf(val) - 1.0f);
        Y[(size_t)row * 64 + col] = val;
      }
    }
  }
}

// ---------- per-node attention scalars: out[n,h] = sum_d Z[n,h,d]*a[h,d] ----------
template<int H, int D>
__global__ void node_scores(const float* __restrict__ Z, const float* __restrict__ avec,
                            float* __restrict__ out, int N) {
  int t = blockIdx.x * blockDim.x + threadIdx.x;
  if (t >= N * H) return;
  int n = t / H, h = t % H;
  const float* z = Z + (size_t)n * (H * D) + h * D;
  const float* a = avec + h * D;
  float s = 0.f;
#pragma unroll
  for (int d = 0; d < D; ++d) s += z[d] * a[d];
  out[t] = s;
}

__global__ void fill_kernel(float* __restrict__ p, float v, int n) {
  int i = blockIdx.x * blockDim.x + threadIdx.x;
  if (i < n) p[i] = v;
}

// ---------- edge pass 1: alpha = leakyrelu(sa[src]+sd[dst]); atomic max per dst ----------
template<int H>
__global__ void edge_alpha_max(const int* __restrict__ edge,
                               const float* __restrict__ sa, const float* __restrict__ sd,
                               float* __restrict__ alpha, float* __restrict__ m, int E) {
  int t = blockIdx.x * blockDim.x + threadIdx.x;
  if (t >= E * H) return;
  int e = t / H, h = t % H;
  int src = edge[e], dst = edge[E + e];
  float a = sa[src * H + h] + sd[dst * H + h];
  a = a > 0.f ? a : 0.2f * a;
  alpha[t] = a;
  atomic_max_f32(&m[dst * H + h], a);
}

// ---------- edge pass 2: e = exp(alpha - m[dst]); atomic sum per dst ----------
template<int H>
__global__ void edge_exp_sum(const int* __restrict__ edge, float* __restrict__ alpha,
                             const float* __restrict__ m, float* __restrict__ s, int E) {
  int t = blockIdx.x * blockDim.x + threadIdx.x;
  if (t >= E * H) return;
  int e = t / H, h = t % H;
  int dst = edge[E + e];
  float ev = expf(alpha[t] - m[dst * H + h]);
  alpha[t] = ev;
  atomic_add_f32(&s[dst * H + h], ev);
}

// ---------- edge pass 3: out[dst,:] += Z[src,:]*w, 16 threads/edge, float4 each ----------
template<int H, int D>
__global__ void edge_aggregate(const int* __restrict__ edge, const float* __restrict__ ebuf,
                               const float* __restrict__ s, const float* __restrict__ Z,
                               float* __restrict__ out, int E) {
  long t = (long)blockIdx.x * blockDim.x + threadIdx.x;
  int e = (int)(t >> 4);
  if (e >= E) return;
  int d0 = ((int)t & 15) * 4;
  int h  = d0 / D;
  int src = edge[e], dst = edge[E + e];
  float w = ebuf[e * H + h] / (s[dst * H + h] + 1e-16f);
  const float4 z = *(const float4*)(Z + (size_t)src * 64 + d0);
  float* op = out + (size_t)dst * 64 + d0;
  atomic_add_f32(op + 0, z.x * w);
  atomic_add_f32(op + 1, z.y * w);
  atomic_add_f32(op + 2, z.z * w);
  atomic_add_f32(op + 3, z.w * w);
}

__global__ void relu_copy_kernel(float* __restrict__ dst, const float* __restrict__ src, size_t n) {
  size_t i = (size_t)blockIdx.x * blockDim.x + threadIdx.x;
  if (i >= n) return;
  float v = src[i];
  dst[i] = v > 0.f ? v : 0.f;
}

// ---------- semantic score: acc += sum_f tanh((o[n]@kw)[f]+kb[f])*q[f], one block/node ----------
__global__ void score_kernel(const float* __restrict__ O, const float* __restrict__ kw,
                             const float* __restrict__ kb, const float* __restrict__ q,
                             float* __restrict__ acc, int N) {
  __shared__ float row[64];
  __shared__ float red[64];
  int n = blockIdx.x, f = threadIdx.x;
  if (n >= N) return;
  row[f] = O[(size_t)n * 64 + f];
  __syncthreads();
  float d = kb[f];
#pragma unroll 8
  for (int k = 0; k < 64; ++k) d += row[k] * kw[k * 64 + f];   // coalesced kw column reads
  red[f] = tanhf(d) * q[f];
  __syncthreads();
  for (int st = 32; st > 0; st >>= 1) {
    if (f < st) red[f] += red[f + st];
    __syncthreads();
  }
  if (f == 0) atomic_add_f32(acc, red[0]);
}

// ---------- beta-softmax blend of the two P-relation outputs ----------
__global__ void combine2_kernel(const float* __restrict__ o0, const float* __restrict__ o1,
                                const float* __restrict__ sc, float invn,
                                float* __restrict__ out, size_t count) {
  size_t i = (size_t)blockIdx.x * blockDim.x + threadIdx.x;
  if (i >= count) return;
  float s0 = sc[0] * invn, s1 = sc[1] * invn;
  float mx = fmaxf(s0, s1);
  float b0 = expf(s0 - mx), b1 = expf(s1 - mx);
  float inv = 1.0f / (b0 + b1);
  out[i] = (b0 * o0[i] + b1 * o1[i]) * inv;
}

// ---------- host-side relation driver ----------
template<int H, int D>
static void run_relation(const int* edge, int E,
                         const float* Zsrc, const float* Zdst, int Nsrc, int Ndst,
                         const float* aSrc, const float* aDst,
                         float* sa, float* sd, float* m, float* ssum, float* alpha,
                         float* out /*pre-zeroed*/, hipStream_t stream) {
  int t1 = Nsrc * H, t2 = Ndst * H, te = E * H;
  node_scores<H, D><<<(t1 + 255) / 256, 256, 0, stream>>>(Zsrc, aSrc, sa, Nsrc);
  node_scores<H, D><<<(t2 + 255) / 256, 256, 0, stream>>>(Zdst, aDst, sd, Ndst);
  fill_kernel<<<(t2 + 255) / 256, 256, 0, stream>>>(m, -INFINITY, t2);
  hipMemsetAsync(ssum, 0, (size_t)t2 * 4, stream);
  edge_alpha_max<H><<<(te + 255) / 256, 256, 0, stream>>>(edge, sa, sd, alpha, m, E);
  edge_exp_sum<H><<<(te + 255) / 256, 256, 0, stream>>>(edge, alpha, m, ssum, E);
  long ta = (long)E * 16;
  edge_aggregate<H, D><<<(int)((ta + 255) / 256), 256, 0, stream>>>(edge, alpha, ssum, Zsrc, out, E);
}

extern "C" void kernel_launch(void* const* d_in, const int* in_sizes, int n_in,
                              void* d_out, int out_size, void* d_ws, size_t ws_size,
                              hipStream_t stream) {
  const float* x_a  = (const float*)d_in[0];
  const float* x_p  = (const float*)d_in[1];
  const float* linw = (const float*)d_in[2];
  const float* linb = (const float*)d_in[3];
  const float* p1w  = (const float*)d_in[4];
  const float* p1b  = (const float*)d_in[5];
  const float* att1 = (const float*)d_in[6];
  const float* k1w  = (const float*)d_in[7];
  const float* k1b  = (const float*)d_in[8];
  const float* q1   = (const float*)d_in[9];
  const float* p2w  = (const float*)d_in[10];
  const float* p2b  = (const float*)d_in[11];
  const float* att2 = (const float*)d_in[12];
  const float* k2w  = (const float*)d_in[13];
  const float* k2b  = (const float*)d_in[14];
  const float* q2   = (const float*)d_in[15];
  const int* e_ap   = (const int*)d_in[16];
  const int* e_pa   = (const int*)d_in[17];
  const int* e_pp   = (const int*)d_in[18];

  const int NA = in_sizes[0] / 128;
  const int NP = in_sizes[1] / 128;
  const int E  = in_sizes[16] / 2;
  const size_t NA64 = (size_t)NA * 64, NP64 = (size_t)NP * 64;

  // ---- workspace layout (floats) ----
  float* ws   = (float*)d_ws;
  float* h_a  = ws;
  float* h_p  = h_a + NA64;
  float* za   = h_p + NP64;
  float* zp   = za + NA64;
  float* oA   = zp + NP64;
  float* oP1  = oA + NA64;
  float* oP2  = oP1 + NP64;
  float* alp  = oP2 + NP64;                 // E*2
  float* sa   = alp + (size_t)E * 2;        // NP*2 (max)
  float* sd   = sa + (size_t)NP * 2;
  float* mbuf = sd + (size_t)NP * 2;
  float* sbuf = mbuf + (size_t)NP * 2;
  float* scAcc = sbuf + (size_t)NP * 2;     // 2 floats (padded)
  unsigned short* pk = (unsigned short*)(scAcc + 64);
  unsigned short* pk_lin0 = pk;             // K=128: 16*512 bf16
  unsigned short* pk_lin1 = pk + 8192;
  unsigned short* pk_p1w0 = pk + 16384;     // K=64: 8*512 bf16
  unsigned short* pk_p1w1 = pk + 20480;
  unsigned short* pk_p2w0 = pk + 24576;
  unsigned short* pk_p2w1 = pk + 28672;

  float* outA = (float*)d_out;
  float* outP = outA + NA64;

  hipMemsetAsync(d_out, 0, (NA64 + NP64) * 4, stream);

  // pack weights into WMMA B layout (bf16)
  pack_w_kernel<<<16, 32, 0, stream>>>(linw,            pk_lin0);
  pack_w_kernel<<<16, 32, 0, stream>>>(linw + 128 * 64, pk_lin1);
  pack_w_kernel<<< 8, 32, 0, stream>>>(p1w,             pk_p1w0);
  pack_w_kernel<<< 8, 32, 0, stream>>>(p1w + 64 * 64,   pk_p1w1);
  pack_w_kernel<<< 8, 32, 0, stream>>>(p2w,             pk_p2w0);
  pack_w_kernel<<< 8, 32, 0, stream>>>(p2w + 64 * 64,   pk_p2w1);

  const int gA = (NA + 63) / 64, gP = (NP + 63) / 64;

  // h = elu(x @ lin_w + lin_b)
  gemm_bias_act<128, 1><<<gA, 128, 0, stream>>>(x_a, pk_lin0, linb,      h_a, NA);
  gemm_bias_act<128, 1><<<gP, 128, 0, stream>>>(x_p, pk_lin1, linb + 64, h_p, NP);

  // ===== HANConv layer 1 (heads=2, D=32) =====
  gemm_bias_act<64, 0><<<gA, 128, 0, stream>>>(h_a, pk_p1w0, p1b,      za, NA);
  gemm_bias_act<64, 0><<<gP, 128, 0, stream>>>(h_p, pk_p1w1, p1b + 64, zp, NP);

  hipMemsetAsync(oP1, 0, NP64 * 4, stream);
  run_relation<2, 32>(e_ap, E, za, zp, NA, NP, att1 + 0,   att1 + 64,  sa, sd, mbuf, sbuf, alp, oP1, stream);
  hipMemsetAsync(oA, 0, NA64 * 4, stream);
  run_relation<2, 32>(e_pa, E, zp, za, NP, NA, att1 + 128, att1 + 192, sa, sd, mbuf, sbuf, alp, oA, stream);
  hipMemsetAsync(oP2, 0, NP64 * 4, stream);
  run_relation<2, 32>(e_pp, E, zp, zp, NP, NP, att1 + 256, att1 + 320, sa, sd, mbuf, sbuf, alp, oP2, stream);

  // new_a = relu(o_a_pa)  (group over 1 relation = identity)
  relu_copy_kernel<<<(int)((NA64 + 255) / 256), 256, 0, stream>>>(h_a, oA, NA64);
  relu_copy_kernel<<<(int)((NP64 + 255) / 256), 256, 0, stream>>>(oP1, oP1, NP64);
  relu_copy_kernel<<<(int)((NP64 + 255) / 256), 256, 0, stream>>>(oP2, oP2, NP64);
  hipMemsetAsync(scAcc, 0, 8, stream);
  score_kernel<<<NP, 64, 0, stream>>>(oP1, k1w, k1b, q1, scAcc + 0, NP);
  score_kernel<<<NP, 64, 0, stream>>>(oP2, k1w, k1b, q1, scAcc + 1, NP);
  combine2_kernel<<<(int)((NP64 + 255) / 256), 256, 0, stream>>>(oP1, oP2, scAcc, 1.0f / NP, h_p, NP64);

  // ===== HANConv layer 2 (heads=1, D=64) =====
  gemm_bias_act<64, 0><<<gA, 128, 0, stream>>>(h_a, pk_p2w0, p2b,      za, NA);
  gemm_bias_act<64, 0><<<gP, 128, 0, stream>>>(h_p, pk_p2w1, p2b + 64, zp, NP);

  hipMemsetAsync(oP1, 0, NP64 * 4, stream);
  run_relation<1, 64>(e_ap, E, za, zp, NA, NP, att2 + 0,   att2 + 64,  sa, sd, mbuf, sbuf, alp, oP1, stream);
  run_relation<1, 64>(e_pa, E, zp, za, NP, NA, att2 + 128, att2 + 192, sa, sd, mbuf, sbuf, alp, outA, stream);  // d_out already zeroed
  hipMemsetAsync(oP2, 0, NP64 * 4, stream);
  run_relation<1, 64>(e_pp, E, zp, zp, NP, NP, att2 + 256, att2 + 320, sa, sd, mbuf, sbuf, alp, oP2, stream);

  relu_copy_kernel<<<(int)((NA64 + 255) / 256), 256, 0, stream>>>(outA, outA, NA64);
  relu_copy_kernel<<<(int)((NP64 + 255) / 256), 256, 0, stream>>>(oP1, oP1, NP64);
  relu_copy_kernel<<<(int)((NP64 + 255) / 256), 256, 0, stream>>>(oP2, oP2, NP64);
  hipMemsetAsync(scAcc, 0, 8, stream);
  score_kernel<<<NP, 64, 0, stream>>>(oP1, k2w, k2b, q2, scAcc + 0, NP);
  score_kernel<<<NP, 64, 0, stream>>>(oP2, k2w, k2b, q2, scAcc + 1, NP);
  combine2_kernel<<<(int)((NP64 + 255) / 256), 256, 0, stream>>>(oP1, oP2, scAcc, 1.0f / NP, outP, NP64);
}
